// NeighborhoodAttention2D_diffusion_20761871909013
// MI455X (gfx1250) — matrix operand order repaired and checked by reading, compiled, not run on hardware
//
#include <hip/hip_runtime.h>
#include <hip/hip_bf16.h>
#include <math.h>

typedef __attribute__((ext_vector_type(16))) _Float16 v16h;
typedef __attribute__((ext_vector_type(8)))  _Float16 v8h;
typedef __attribute__((ext_vector_type(8)))  float    v8f;

#define BB    2
#define HH    64
#define WW    64
#define CC    128
#define NHEAD 4
#define HD    32
#define KW    7
#define TOKENS (BB*HH*WW)   // 8192

union V16H { v16h v; v8h h[2]; };

// ---------------------------------------------------------------------------
// Kernel 1: t -> sinusoidal emb -> silu -> [128x256] matvec -> scale/shift
// ---------------------------------------------------------------------------
__global__ void ada_kernel(const int* __restrict__ t,
                           const float* __restrict__ ln_w,   // [128,256]
                           const float* __restrict__ ln_b,   // [256]
                           float* __restrict__ ss) {         // [2,256]
    __shared__ float se[2][128];
    int tid = threadIdx.x;                 // 256 threads
    {
        int b = tid >> 7, i = tid & 127;
        float tf = (float)t[b] * (1.0f / 100.0f) * 4000.0f;
        const float lf = -logf(10000.0f) / 63.0f;
        float e = (i < 64) ? sinf(tf * expf((float)i * lf))
                           : cosf(tf * expf((float)(i - 64) * lf));
        se[b][i] = e / (1.0f + expf(-e));  // silu
    }
    __syncthreads();
    int j = tid;                           // output column 0..255
    for (int b = 0; b < 2; ++b) {
        float acc = ln_b[j];
        for (int i = 0; i < 128; ++i) acc += se[b][i] * ln_w[i * 256 + j];
        ss[b * 256 + j] = acc;
    }
}

// ---------------------------------------------------------------------------
// Kernel 2: per-token LayerNorm + AdaLN modulation, f32 -> f16
// ---------------------------------------------------------------------------
__global__ void ln_mod_kernel(const float* __restrict__ x,   // [8192,128]
                              const float* __restrict__ ss,  // [2,256]
                              _Float16* __restrict__ y) {    // [8192,128]
    __shared__ float s1[128], s2[128];
    int tok = blockIdx.x;
    int c   = threadIdx.x;                 // 128 threads
    int b   = tok >> 12;
    float v = x[tok * 128 + c];
    s1[c] = v; s2[c] = v * v;
    __syncthreads();
    for (int off = 64; off > 0; off >>= 1) {
        if (c < off) { s1[c] += s1[c + off]; s2[c] += s2[c + off]; }
        __syncthreads();
    }
    float mu  = s1[0] * (1.0f / 128.0f);
    float var = s2[0] * (1.0f / 128.0f) - mu * mu;
    float inv = rsqrtf(var + 1e-5f);
    float sc  = ss[b * 256 + c];
    float sh  = ss[b * 256 + 128 + c];
    float yv  = (v - mu) * inv * (1.0f + sc) + sh;
    y[tok * 128 + c] = (_Float16)yv;
}

// ---------------------------------------------------------------------------
// Kernel 3/5: WMMA GEMM  out[M,N] = A[M,128](f16) * Bw[128,N](f32) + bias[N]
// Block: 256 threads (8 waves). Block tile 64 rows x 32 cols; each wave owns
// one 16x16 tile; K=128 walked as 4x v_wmma_f32_16x16x32_f16.
// A tile staged via CDNA5 async global->LDS DMA (ASYNCcnt tracked).
// ---------------------------------------------------------------------------
template<int N>
__global__ __launch_bounds__(256)
void gemm_wmma(const _Float16* __restrict__ A,
               const float* __restrict__ Bw,
               const float* __restrict__ bias,
               float* __restrict__ out) {
    __shared__ __attribute__((aligned(16))) _Float16 As[64][136]; // padded
    __shared__ __attribute__((aligned(16))) _Float16 Bs[32][136]; // transposed [n][k]
    int tid = threadIdx.x;
    int rowBase = blockIdx.x * 64;
    int colBase = blockIdx.y * 32;

    // Stage A: 64x128 halves, async 16B global->LDS copies (no VGPR bounce)
    #pragma unroll
    for (int i = 0; i < 4; ++i) {
        int cidx = tid + i * 256;          // 0..1023 chunks of 8 halves
        int r = cidx >> 4;
        int o = (cidx & 15) * 8;
        const _Float16* g = &A[(rowBase + r) * 128 + o];
        unsigned int lo = (unsigned int)(unsigned long long)&As[r][o]; // LDS offset (addr[31:0])
        asm volatile("global_load_async_to_lds_b128 %0, %1, off"
                     :: "v"(lo), "v"(g) : "memory");
    }
    // Stage B transposed with f32->f16 convert (needs VALU anyway)
    #pragma unroll
    for (int i = 0; i < 16; ++i) {
        int e = tid + i * 256;             // 0..4095
        int k = e >> 5;
        int n = e & 31;
        Bs[n][k] = (_Float16)Bw[k * N + colBase + n];
    }
    asm volatile("s_wait_asynccnt 0x0" ::: "memory");
    __syncthreads();

    int wv = tid >> 5, lane = tid & 31;
    int tr = wv >> 1, tc = wv & 1;         // 4 row tiles x 2 col tiles
    int half = lane >> 4, l15 = lane & 15;

    v8f acc = {};
    #pragma unroll
    for (int kk = 0; kk < 4; ++kk) {
        V16H a, b;
        // A frag: lane<16 -> K{0..7,16..23}; lane>=16 -> K{8..15,24..31}
        const _Float16* ap = &As[tr * 16 + l15][kk * 32 + half * 8];
        a.h[0] = *(const v8h*)(ap);
        a.h[1] = *(const v8h*)(ap + 16);
        // B frag: lane<16 -> K{0..7,16..23}; lane>=16 -> K{8..15,24..31} (column = l15)
        const _Float16* bp = &Bs[tc * 16 + l15][kk * 32 + half * 8];
        b.h[0] = *(const v8h*)(bp);
        b.h[1] = *(const v8h*)(bp + 16);
        acc = __builtin_amdgcn_wmma_f32_16x16x32_f16(
                  false, a.v, false, b.v, (short)0, acc, false, false);
    }

    int col = colBase + tc * 16 + l15;
    float bv = bias[col];
    #pragma unroll
    for (int r = 0; r < 8; ++r) {
        int row = rowBase + tr * 16 + half * 8 + r;
        out[row * N + col] = acc[r] + bv;
    }
}

// ---------------------------------------------------------------------------
// Kernel 4: 7x7 neighborhood attention. Block = (b, head, h); stages the
// 7x64x32 K and V slabs (f16) + rpb slice into LDS once (coalesced), then
// one wave per query: lane j holds score for key j (and j+32), softmax via
// wave shuffles, lane d accumulates the 49-key V sum from LDS.
// ---------------------------------------------------------------------------
__global__ __launch_bounds__(256)
void natten_kernel(const float* __restrict__ qkv,   // [8192,384] (q|k|v per head)
                   const float* __restrict__ rpb,   // [4,13,13]
                   _Float16* __restrict__ o) {      // [8192,128]
    __shared__ _Float16 Ks[KW][WW][HD];   // 28 KB
    __shared__ _Float16 Vs[KW][WW][HD];   // 28 KB
    __shared__ float    rps[169];

    int blk  = blockIdx.x;                 // b*NHEAD*HH + head*HH + h
    int b    = blk / (NHEAD * HH);
    int rem  = blk % (NHEAD * HH);
    int head = rem / HH;
    int h    = rem % HH;
    int tid  = threadIdx.x;
    int wv   = tid >> 5, lane = tid & 31;

    int hstart = h - 3; if (hstart < 0) hstart = 0; if (hstart > HH - KW) hstart = HH - KW;

    // ---- cooperative staging (coalesced over d) ----
    for (int i = tid; i < KW * WW * HD; i += 256) {
        int d = i & 31, wcol = (i >> 5) & 63, r = i >> 11;
        size_t base = (size_t)((b * HH + hstart + r) * WW + wcol) * 384 + head * 32 + d;
        Ks[r][wcol][d] = (_Float16)qkv[base + 128];
        Vs[r][wcol][d] = (_Float16)qkv[base + 256];
    }
    for (int i = tid; i < 169; i += 256) rps[i] = rpb[head * 169 + i];
    __syncthreads();

    const float qscale = 0.1767766952966369f;      // 32^-0.5

    for (int qi = 0; qi < 8; ++qi) {
        int w = wv * 8 + qi;
        int wstart = w - 3; if (wstart < 0) wstart = 0; if (wstart > WW - KW) wstart = WW - KW;
        int tq = (b * HH + h) * WW + w;
        float qd = qkv[(size_t)tq * 384 + head * 32 + lane] * qscale;

        // per-lane keys: j0 = lane (always < 49), j1 = lane + 32
        int j0 = lane;
        int r0 = j0 / 7, kw0 = wstart + j0 % 7;
        int j1 = lane + 32;
        bool v1 = (j1 < 49);
        int r1  = v1 ? j1 / 7 : r0;
        int kw1 = v1 ? wstart + j1 % 7 : kw0;

        const _Float16* kl0 = &Ks[r0][kw0][0];
        const _Float16* kl1 = &Ks[r1][kw1][0];

        float s0 = rps[(hstart + r0 - h + 6) * 13 + (kw0 - w + 6)];
        float s1 = v1 ? rps[(hstart + r1 - h + 6) * 13 + (kw1 - w + 6)] : -1e30f;
        #pragma unroll
        for (int d = 0; d < 32; ++d) {
            float qv = __shfl(qd, d, 32);
            s0 += qv * (float)kl0[d];
            s1 += qv * (float)kl1[d];      // invalid lanes stay ~-1e30
        }

        // softmax over 49 (s0 all lanes, s1 lanes<17)
        float m = fmaxf(s0, s1);
        for (int off = 16; off > 0; off >>= 1) m = fmaxf(m, __shfl_xor(m, off, 32));
        float e0 = __expf(s0 - m);
        float e1 = v1 ? __expf(s1 - m) : 0.0f;
        float sum = e0 + e1;
        for (int off = 16; off > 0; off >>= 1) sum += __shfl_xor(sum, off, 32);
        float inv = 1.0f / sum;
        float p0 = e0 * inv, p1 = e1 * inv;

        // out[d] = sum_j p_j * v[key_j][d]   (lane = d)
        float accum = 0.0f;
        for (int j = 0; j < 49; ++j) {
            float pj = (j < 32) ? __shfl(p0, j, 32) : __shfl(p1, j - 32, 32);
            accum += pj * (float)Vs[j / 7][wstart + j % 7][lane];
        }
        o[(size_t)tq * 128 + head * 32 + lane] = (_Float16)accum;
    }
}

// ---------------------------------------------------------------------------
// Launch
// ---------------------------------------------------------------------------
extern "C" void kernel_launch(void* const* d_in, const int* in_sizes, int n_in,
                              void* d_out, int out_size, void* d_ws, size_t ws_size,
                              hipStream_t stream) {
    (void)in_sizes; (void)n_in; (void)out_size; (void)ws_size;
    const float* x      = (const float*)d_in[0];
    // d_in[1] = cond, unused by the reference
    const int*   t      = (const int*)d_in[2];
    const float* ln_w   = (const float*)d_in[3];
    const float* ln_b   = (const float*)d_in[4];
    const float* qkv_w  = (const float*)d_in[5];
    const float* qkv_b  = (const float*)d_in[6];
    const float* rpb    = (const float*)d_in[7];
    const float* proj_w = (const float*)d_in[8];
    const float* proj_b = (const float*)d_in[9];
    float* out = (float*)d_out;

    char* ws = (char*)d_ws;
    float*     ss     = (float*)ws;                                  // 2 KB
    _Float16*  y_h    = (_Float16*)(ws + 2048);                      // 2 MB
    _Float16*  attn_h = (_Float16*)(ws + 2048 + 2097152);            // 2 MB
    float*     qkvbuf = (float*)(ws + 2048 + 2 * 2097152);           // 12 MB

    ada_kernel   <<<1,    256, 0, stream>>>(t, ln_w, ln_b, ss);
    ln_mod_kernel<<<TOKENS, 128, 0, stream>>>(x, ss, y_h);
    gemm_wmma<384><<<dim3(TOKENS / 64, 12), 256, 0, stream>>>(y_h, qkv_w, qkv_b, qkvbuf);
    natten_kernel<<<BB * NHEAD * HH, 256, 0, stream>>>(qkvbuf, rpb, attn_h);
    gemm_wmma<128><<<dim3(TOKENS / 64, 4), 256, 0, stream>>>(attn_h, proj_w, proj_b, out);
}
